// LinearRNNLoop_18287970746949
// MI455X (gfx1250) — matrix operand order repaired and checked by reading, not compiled, not run on hardware
//
#include <hip/hip_runtime.h>

typedef __attribute__((ext_vector_type(16))) __bf16 v16bf;
typedef __attribute__((ext_vector_type(8)))  float  v8f;

namespace {
constexpr int Bsz = 32;
constexpr int Ssz = 512;
constexpr int Isz = 1024;
constexpr int Hsz = 1024;
constexpr int Msz = Bsz * Ssz;                         // 16384 rows of x_proj
constexpr long long SeqStride = (long long)Ssz * Hsz;  // batch stride in d_out
}

__device__ __forceinline__ unsigned int f2bf(float f) {
  // round-to-nearest-even f32 -> bf16 (bits in low 16)
  unsigned int u = __float_as_uint(f);
  u += 0x7FFFu + ((u >> 16) & 1u);
  return u >> 16;
}

union FragBF {
  uint4 q[2];            // two b128 loads
  unsigned short u[16];  // element view
  v16bf v;               // WMMA operand view
};

// ---------------------------------------------------------------- conversions
__global__ __launch_bounds__(256)
void cvt_f32_to_bf16(const float* __restrict__ src,
                     unsigned short* __restrict__ dst, int n) {
  int i = (blockIdx.x * blockDim.x + threadIdx.x) * 8;
  if (i >= n) return;
  float4 a = *(const float4*)(src + i);
  float4 b = *(const float4*)(src + i + 4);
  uint4 o;
  o.x = f2bf(a.x) | (f2bf(a.y) << 16);
  o.y = f2bf(a.z) | (f2bf(a.w) << 16);
  o.z = f2bf(b.x) | (f2bf(b.y) << 16);
  o.w = f2bf(b.z) | (f2bf(b.w) << 16);
  *(uint4*)(dst + i) = o;
}

// ------------------------------------------------- phase 1: x_proj big GEMM
// out[m,n] = sum_k Xbf[m,k] * Wbf[n,k] + bias_ih[n] + bias_hh[n]
// Block = 8 waves: block tile 128(M) x 64(N).
// B tile (64 rows x 32 K, 4KB bf16) staged in LDS, double-buffered, shared by
// all 8 waves (8x L2 traffic reduction). A streamed register-direct with a
// double-buffered fragment pipeline. All 8 ds_loads issued before the 4 WMMAs
// (distinct fragment registers) so dscnt waits are incremental, not 0.
__global__ __launch_bounds__(256)
void xproj_gemm(const unsigned short* __restrict__ X,   // [M, I] bf16
                const unsigned short* __restrict__ W,   // [H, I] bf16
                const float* __restrict__ bias_ih,
                const float* __restrict__ bias_hh,
                float* __restrict__ out) {              // [M, H] f32
  constexpr int KSTEPS = Isz / 32;                      // 32 K-blocks
  __shared__ unsigned short ldsB[2][64 * 32];           // 2 x 4KB

  const int waveId = (int)(threadIdx.x >> 5);           // 0..7
  const int lane   = (int)(threadIdx.x & 31);
  const int nb     = (int)(blockIdx.x & 15);            // 16 N-strips of 64
  const int mblk   = (int)(blockIdx.x >> 4);            // 128 M-blocks of 128
  const int l15 = lane & 15;
  const int kh  = lane >> 4;

  // ---- B staging assignment: 256 threads x 16B = 4KB tile
  const int brow = (int)(threadIdx.x >> 2);             // 0..63
  const int bc4  = (int)((threadIdx.x & 3) * 8);        // element offset 0/8/16/24
  const unsigned short* gB = W + (size_t)(nb * 64 + brow) * Isz + bc4;

  // ---- A row for this wave (m-tile = mblk*8 + waveId)
  const unsigned short* xrow =
      X + (size_t)((mblk * 8 + waveId) * 16 + l15) * Isz;

  v8f c0 = {}, c1 = {}, c2 = {}, c3 = {};

  // prologue: stage B(k=0), load A(k=0)
  {
    uint4 s = *(const uint4*)(gB);
    *(uint4*)(&ldsB[0][brow * 32 + bc4]) = s;
  }
  FragBF Acur;
  Acur.q[0] = *(const uint4*)(xrow + 0 + kh * 16);
  Acur.q[1] = *(const uint4*)(xrow + 8 + kh * 16);
  __syncthreads();

  // per-wave LDS read base (element offsets within a buffer)
  const int lds0 = (0 * 16 + l15) * 32 + kh * 16;
  const int lds1 = (1 * 16 + l15) * 32 + kh * 16;
  const int lds2 = (2 * 16 + l15) * 32 + kh * 16;
  const int lds3 = (3 * 16 + l15) * 32 + kh * 16;

#pragma unroll 1
  for (int it = 0; it < KSTEPS - 1; ++it) {
    const int k0n = (it + 1) * 32;
    // prefetch next-iteration globals while we compute
    uint4 sB = *(const uint4*)(gB + k0n);
    FragBF Anxt;
    Anxt.q[0] = *(const uint4*)(xrow + k0n + kh * 16);
    Anxt.q[1] = *(const uint4*)(xrow + k0n + 8 + kh * 16);

    // compute on current LDS buffer: issue all 8 ds_loads, then 4 WMMAs
    {
      const unsigned short* lb = ldsB[it & 1];
      FragBF Bf0, Bf1, Bf2, Bf3;
      Bf0.q[0] = *(const uint4*)(lb + lds0);
      Bf0.q[1] = *(const uint4*)(lb + lds0 + 8);
      Bf1.q[0] = *(const uint4*)(lb + lds1);
      Bf1.q[1] = *(const uint4*)(lb + lds1 + 8);
      Bf2.q[0] = *(const uint4*)(lb + lds2);
      Bf2.q[1] = *(const uint4*)(lb + lds2 + 8);
      Bf3.q[0] = *(const uint4*)(lb + lds3);
      Bf3.q[1] = *(const uint4*)(lb + lds3 + 8);
      c0 = __builtin_amdgcn_wmma_f32_16x16x32_bf16(false, Acur.v, false, Bf0.v, (short)0, c0, false, false);
      c1 = __builtin_amdgcn_wmma_f32_16x16x32_bf16(false, Acur.v, false, Bf1.v, (short)0, c1, false, false);
      c2 = __builtin_amdgcn_wmma_f32_16x16x32_bf16(false, Acur.v, false, Bf2.v, (short)0, c2, false, false);
      c3 = __builtin_amdgcn_wmma_f32_16x16x32_bf16(false, Acur.v, false, Bf3.v, (short)0, c3, false, false);
    }

    __syncthreads();   // all waves done reading buffer (it+1)&1 from iter it-1
    *(uint4*)(&ldsB[(it + 1) & 1][brow * 32 + bc4]) = sB;
    __syncthreads();   // staged data visible to all waves
    Acur = Anxt;
  }

  // final compute iteration (buffer (KSTEPS-1)&1)
  {
    const unsigned short* lb = ldsB[(KSTEPS - 1) & 1];
    FragBF Bf0, Bf1, Bf2, Bf3;
    Bf0.q[0] = *(const uint4*)(lb + lds0);
    Bf0.q[1] = *(const uint4*)(lb + lds0 + 8);
    Bf1.q[0] = *(const uint4*)(lb + lds1);
    Bf1.q[1] = *(const uint4*)(lb + lds1 + 8);
    Bf2.q[0] = *(const uint4*)(lb + lds2);
    Bf2.q[1] = *(const uint4*)(lb + lds2 + 8);
    Bf3.q[0] = *(const uint4*)(lb + lds3);
    Bf3.q[1] = *(const uint4*)(lb + lds3 + 8);
    c0 = __builtin_amdgcn_wmma_f32_16x16x32_bf16(false, Acur.v, false, Bf0.v, (short)0, c0, false, false);
    c1 = __builtin_amdgcn_wmma_f32_16x16x32_bf16(false, Acur.v, false, Bf1.v, (short)0, c1, false, false);
    c2 = __builtin_amdgcn_wmma_f32_16x16x32_bf16(false, Acur.v, false, Bf2.v, (short)0, c2, false, false);
    c3 = __builtin_amdgcn_wmma_f32_16x16x32_bf16(false, Acur.v, false, Bf3.v, (short)0, c3, false, false);
  }

  // C/D layout: lane n = lane&15; VGPR r -> m = r + (lane>>4)*8
  const int ncol = nb * 64 + l15;
  const float bb0 = bias_ih[ncol +  0] + bias_hh[ncol +  0];
  const float bb1 = bias_ih[ncol + 16] + bias_hh[ncol + 16];
  const float bb2 = bias_ih[ncol + 32] + bias_hh[ncol + 32];
  const float bb3 = bias_ih[ncol + 48] + bias_hh[ncol + 48];
  float* obase = out + (size_t)((mblk * 8 + waveId) * 16 + kh * 8) * Hsz + ncol;
#pragma unroll
  for (int r = 0; r < 8; ++r) {
    size_t off = (size_t)r * Hsz;
    obase[off +  0] = c0[r] + bb0;
    obase[off + 16] = c1[r] + bb1;
    obase[off + 32] = c2[r] + bb2;
    obase[off + 48] = c3[r] + bb3;
  }
}

// ------------------------------------------------- phase 2: one recurrence step
// outT (in d_out, already holds x_proj[:,t,:]) += hprev_bf16 @ Whh^T.
// hprev is a bf16 [B,H] ping-pong buffer; result also written back as bf16 to
// hnext for the next step (no per-iteration f32->bf16 in the K-loop).
// Register double-buffered pipeline (2 K-blocks in flight).
__global__ __launch_bounds__(256)
void rnn_step(const unsigned short* __restrict__ hprev, // [B,H] bf16
              const unsigned short* __restrict__ Whh,   // [H,H] bf16
              float* __restrict__ outT,                 // row b at outT + b*SeqStride
              unsigned short* __restrict__ hnext) {     // [B,H] bf16
  const int wave = (int)((blockIdx.x * blockDim.x + threadIdx.x) >> 5);
  const int lane = (int)(threadIdx.x & 31);
  const int mt = wave >> 6;             // 0..1 (B=32 rows)
  const int nt = wave & 63;             // 64 N-tiles
  const int l15 = lane & 15;
  const int kh  = lane >> 4;

  const unsigned short* hrow = hprev + (size_t)(mt * 16 + l15) * Hsz;
  const unsigned short* wrow = Whh + (size_t)(nt * 16 + l15) * Hsz;

#define LOAD_AB(Af, Bf, kk)                                        \
  do {                                                             \
    Af.q[0] = *(const uint4*)(hrow + (kk) + kh * 16);              \
    Af.q[1] = *(const uint4*)(hrow + (kk) + kh * 16 + 8);          \
    Bf.q[0] = *(const uint4*)(wrow + (kk) + kh * 16);              \
    Bf.q[1] = *(const uint4*)(wrow + (kk) + kh * 16 + 8);          \
  } while (0)

  v8f c = {};
  FragBF A0, B0, A1, B1;
  LOAD_AB(A0, B0, 0);
#pragma unroll 1
  for (int k0 = 0; k0 < Hsz - 64; k0 += 64) {
    LOAD_AB(A1, B1, k0 + 32);
    c = __builtin_amdgcn_wmma_f32_16x16x32_bf16(false, A0.v, false, B0.v, (short)0, c, false, false);
    LOAD_AB(A0, B0, k0 + 64);
    c = __builtin_amdgcn_wmma_f32_16x16x32_bf16(false, A1.v, false, B1.v, (short)0, c, false, false);
  }
  LOAD_AB(A1, B1, Hsz - 32);
  c = __builtin_amdgcn_wmma_f32_16x16x32_bf16(false, A0.v, false, B0.v, (short)0, c, false, false);
  c = __builtin_amdgcn_wmma_f32_16x16x32_bf16(false, A1.v, false, B1.v, (short)0, c, false, false);
#undef LOAD_AB

  // C/D: lane n = lane&15; VGPR r -> m = r + (lane>>4)*8
  const int ncol = nt * 16 + l15;
  const int mrow = mt * 16 + kh * 8;
  float* o = outT + (size_t)mrow * SeqStride + ncol;
  unsigned short* hn = hnext + (size_t)mrow * Hsz + ncol;
#pragma unroll
  for (int r = 0; r < 8; ++r) {
    float v = o[(size_t)r * SeqStride] + c[r];  // fused + x_proj (in place)
    o[(size_t)r * SeqStride] = v;
    hn[(size_t)r * Hsz] = (unsigned short)f2bf(v);
  }
}

// ------------------------------------------------- h_last gather
__global__ __launch_bounds__(256)
void copy_hlast(const float* __restrict__ seqLast,  // d_out + (S-1)*H
                float* __restrict__ dst) {          // d_out + B*S*H
  int i = (int)(blockIdx.x * blockDim.x + threadIdx.x);
  int b = i >> 10;            // / H (H = 1024)
  int n = i & (Hsz - 1);
  dst[i] = seqLast[(size_t)b * SeqStride + n];
}

extern "C" void kernel_launch(void* const* d_in, const int* in_sizes, int n_in,
                              void* d_out, int out_size, void* d_ws, size_t ws_size,
                              hipStream_t stream) {
  (void)in_sizes; (void)n_in; (void)out_size; (void)ws_size;
  const float* inputs  = (const float*)d_in[0];   // [B,S,I]
  const float* hidden  = (const float*)d_in[1];   // [1,B,H]
  const float* w_ih    = (const float*)d_in[2];   // [H,I]
  const float* bias_ih = (const float*)d_in[3];   // [H]
  const float* w_hh    = (const float*)d_in[4];   // [H,H]
  const float* bias_hh = (const float*)d_in[5];   // [H]
  float* out = (float*)d_out;                     // [B,S,H] then [B,H]

  // workspace: bf16 inputs (32MB) + bf16 W_ih (2MB) + bf16 W_hh (2MB)
  //            + bf16 h ping-pong (2 x 64KB)
  char* ws = (char*)d_ws;
  unsigned short* xbf  = (unsigned short*)ws;
  unsigned short* wihb = (unsigned short*)(ws + (size_t)Msz * Isz * 2);
  unsigned short* whhb = (unsigned short*)(ws + (size_t)Msz * Isz * 2 + (size_t)Hsz * Isz * 2);
  unsigned short* hbuf0 = (unsigned short*)(ws + (size_t)Msz * Isz * 2 + (size_t)Hsz * Isz * 2
                                               + (size_t)Hsz * Hsz * 2);
  unsigned short* hbuf1 = hbuf0 + (size_t)Bsz * Hsz;
  unsigned short* hb[2] = {hbuf0, hbuf1};

  // f32 -> bf16 conversions
  cvt_f32_to_bf16<<<(Msz * Isz / 8) / 256, 256, 0, stream>>>(inputs, xbf, Msz * Isz);
  cvt_f32_to_bf16<<<(Hsz * Isz / 8) / 256, 256, 0, stream>>>(w_ih, wihb, Hsz * Isz);
  cvt_f32_to_bf16<<<(Hsz * Hsz / 8) / 256, 256, 0, stream>>>(w_hh, whhb, Hsz * Hsz);
  cvt_f32_to_bf16<<<(Bsz * Hsz / 8) / 256, 256, 0, stream>>>(hidden, hbuf0, Bsz * Hsz);

  // phase 1: x_proj (+biases) straight into d_out's seq region
  {
    const int blocks = (Msz / 128) * (Hsz / 64);   // 128 x 16 = 2048
    xproj_gemm<<<blocks, 256, 0, stream>>>(xbf, wihb, bias_ih, bias_hh, out);
  }

  // phase 2: 512 sequential in-place steps with bf16 h ping-pong
  {
    const int waves  = (Bsz / 16) * (Hsz / 16);    // 128 waves
    const int blocks = waves * 32 / 256;           // 16 blocks
    for (int t = 0; t < Ssz; ++t) {
      rnn_step<<<blocks, 256, 0, stream>>>(hb[t & 1], whhb,
                                           out + (size_t)t * Hsz, hb[(t + 1) & 1]);
    }
  }

  // h_last = seq[:, S-1, :]
  copy_hlast<<<(Bsz * Hsz) / 256, 256, 0, stream>>>(out + (size_t)(Ssz - 1) * Hsz,
                                                    out + (size_t)Bsz * Ssz * Hsz);
}
